// SineLayer_LoE_34754875359890
// MI455X (gfx1250) — matrix operand order repaired and checked by reading, not compiled, not run on hardware
//
#include <hip/hip_runtime.h>
#include <math.h>
#include <stdint.h>

typedef __bf16 bf16_t;
typedef bf16_t v16bf __attribute__((ext_vector_type(16)));
typedef float  v8f   __attribute__((ext_vector_type(8)));

#define BATCH  65536
#define CIN    512
#define COUT   512
#define NEXP   4
#define OMEGA0 30.0f
#define ASCALE 16.0f      // 2**(5-1)
#define MTILE  64         // rows per block (four 16-row WMMA sub-tiles)
#define TPB    512        // 16 wave32s
#define NTW    8          // N-tiles per wave: 4 wave-groups * 8 * 16 = 512 cols

// dynamic-LDS byte layout
#define LDS_AH    0            // 64 KB : A hi  [p][kc][lane][16]
#define LDS_AL    65536        // 64 KB : A lo
#define LDS_B     131072       // 2 x 64 KB : per-buffer [Bh 32KB | Bl 32KB], [nt][lane][16]
#define LDS_ROWS  262144       // 256 B
#define LDS_TOTAL (262144 + 256)

// ---------------- expert bucketing ----------------
__global__ void zero_cnt_kernel(int* cnt) {
  if (threadIdx.x < NEXP) cnt[threadIdx.x] = 0;
}

__global__ void build_perm_kernel(const float* __restrict__ coords,
                                  int* __restrict__ cnt, int* __restrict__ perm) {
  int b = blockIdx.x * blockDim.x + threadIdx.x;
  if (b >= BATCH) return;
  float x = coords[2 * b], y = coords[2 * b + 1];
  int xi = ((int)floorf(x * ASCALE)) & 1;       // coords in [0,1) -> aff >= 0
  int yi = ((int)floorf(y * ASCALE)) & 1;
  int t = 2 * xi + yi;
  int slot = atomicAdd(&cnt[t], 1);
  perm[t * BATCH + slot] = b;
}

// ---------------- W -> bf16 hi/lo, pre-swizzled into WMMA B-operand layout ----
// B tile 32(K)x16(N), this code's k order (A same): lanes 0-15 hold K=0..15 for N=lane, lanes 16-31 hold
// K=16..31 for N=lane-16; 16 bf16 contiguous per lane (VGPR r = K pair 2r,2r+1).
__global__ void convert_w_kernel(const float* __restrict__ W,
                                 bf16_t* __restrict__ Bh, bf16_t* __restrict__ Bl) {
  int idx = blockIdx.x * blockDim.x + threadIdx.x;   // [e][n][k] row-major
  int e = idx >> 18;
  int n = (idx >> 9) & 511;
  int k = idx & 511;
  float w = W[idx];
  bf16_t h = (bf16_t)w;
  bf16_t l = (bf16_t)(w - (float)h);
  int nt = n >> 4, nl = n & 15;
  int kc = k >> 5, kk = k & 31;
  int lane = (kk < 16) ? nl : (nl + 16);
  int j = kk & 15;
  int o = (((e * 32 + nt) * 16 + kc) * 32 + lane) * 16 + j;
  Bh[o] = h;
  Bl[o] = l;
}

// ---------------- gather-GEMM + sin, per expert ----------------
__global__ __launch_bounds__(TPB)
void moe_siren_gemm_kernel(const float* __restrict__ feats,
                           const bf16_t* __restrict__ Bh,
                           const bf16_t* __restrict__ Bl,
                           const int* __restrict__ perm,
                           const int* __restrict__ cnt,
                           float* __restrict__ out) {
  extern __shared__ __align__(32) char smem[];
  bf16_t* ldsAh = (bf16_t*)(smem + LDS_AH);
  bf16_t* ldsAl = (bf16_t*)(smem + LDS_AL);
  int* rowsS = (int*)(smem + LDS_ROWS);

  const int e = blockIdx.y;
  const int count = cnt[e];
  const int base = blockIdx.x * MTILE;
  if (base >= count) return;
  const int tid = threadIdx.x;

  if (tid < MTILE) {
    int slot = base + tid;
    rowsS[tid] = (slot < count) ? perm[e * BATCH + slot] : -1;
  }
  __syncthreads();

  // ---- stage A: 64 gathered f32 rows -> bf16 hi/lo, scattered into WMMA-A layout.
  // A layout (this code's contiguous k order, same as B): lanes 0-15 = M, K {0..15}; lanes 16-31 K {16..31}.
  {
    int r = tid >> 3;                 // 0..63, 8 threads per row
    int k0 = (tid & 7) * 64;          // 64 contiguous cin per thread (coalesced b128s)
    int row = rowsS[r];
    const float* src = feats + (long)(row < 0 ? 0 : row) * CIN;
    int p = r >> 4, rr = r & 15;
#pragma unroll 4
    for (int i = 0; i < 64; ++i) {
      int k = k0 + i;
      float w = (row < 0) ? 0.0f : src[k];
      bf16_t h = (bf16_t)w;
      bf16_t l = (bf16_t)(w - (float)h);
      int kc = k >> 5, kk = k & 31, g = kk >> 3;
      int lane2 = rr + ((g & 2) ? 16 : 0);
      int j = kk & 15;
      int o = ((p * 16 + kc) * 32 + lane2) * 16 + j;
      ldsAh[o] = h;
      ldsAl[o] = l;
    }
  }

  // ---- async B staging (double-buffered): 64 KB per K-chunk, once per block,
  // shared by all 4 M-sub-tiles. LDS offset = low 32 bits of flat shared addr.
  const unsigned smem_base = (unsigned)(uintptr_t)smem;
  const unsigned long long bh_g = (unsigned long long)(uintptr_t)Bh;
  const unsigned long long bl_g = (unsigned long long)(uintptr_t)Bl;

  auto stageB = [&](int kc) {
    unsigned lbase = smem_base + LDS_B + ((unsigned)(kc & 1) << 16);
#pragma unroll
    for (int i = 0; i < 4; ++i) {
      unsigned c = (unsigned)tid * 4u + (unsigned)i;           // 0..2047 16B chunks
      unsigned nt = c >> 6;                                    // 1 KB per tile
      unsigned gb = ((((unsigned)e * 32u + nt) * 16u + (unsigned)kc) << 10) + ((c & 63u) << 4);
      unsigned lh = lbase + (c << 4);
      unsigned ll = lh + 32768u;
      unsigned long long gh = bh_g + gb;
      unsigned long long gl = bl_g + gb;
      asm volatile("global_load_async_to_lds_b128 %0, %1, off"
                   :: "v"(lh), "v"(gh) : "memory");
      asm volatile("global_load_async_to_lds_b128 %0, %1, off"
                   :: "v"(ll), "v"(gl) : "memory");
    }
  };

  stageB(0);

  const int wave = tid >> 5;            // 16 wave32s
  const int lane = tid & 31;
  const int p = wave >> 2;              // 16-row sub-tile 0..3
  const int nbase = (wave & 3) * NTW;   // 8 N-tiles per wave

  v8f acc[NTW] = {};

  for (int kc = 0; kc < 16; ++kc) {
    asm volatile("s_wait_asynccnt 0x0" ::: "memory");  // own async issues landed
    __syncthreads();                                   // => everyone's landed
    if (kc < 15) stageB(kc + 1);                       // prefetch next chunk

    const v16bf ah = *(const v16bf*)&ldsAh[((p * 16 + kc) * 32 + lane) * 16];
    const v16bf al = *(const v16bf*)&ldsAl[((p * 16 + kc) * 32 + lane) * 16];
    const bf16_t* bbuf = (const bf16_t*)(smem + LDS_B + ((kc & 1) << 16));
#pragma unroll
    for (int q = 0; q < NTW; ++q) {
      int nt = nbase + q;
      const v16bf bh = *(const v16bf*)&bbuf[(nt * 32 + lane) * 16];
      const v16bf bl = *(const v16bf*)&bbuf[16384 + (nt * 32 + lane) * 16];
      // fp32-ish product via bf16 split: a*w ~= al*bh + ah*bl + ah*bh
      acc[q] = __builtin_amdgcn_wmma_f32_16x16x32_bf16(false, al, false, bh,
                                                       (short)0, acc[q], false, false);
      acc[q] = __builtin_amdgcn_wmma_f32_16x16x32_bf16(false, ah, false, bl,
                                                       (short)0, acc[q], false, false);
      acc[q] = __builtin_amdgcn_wmma_f32_16x16x32_bf16(false, ah, false, bh,
                                                       (short)0, acc[q], false, false);
    }
  }

  // C/D layout: VGPR v, lanes 0-15 -> (M=v, N=lane); lanes 16-31 -> (M=v+8, N=lane-16)
  const int mhalf = (lane >> 4) * 8;
  const int ncol = lane & 15;
#pragma unroll
  for (int q = 0; q < NTW; ++q) {
    int nt = nbase + q;
#pragma unroll
    for (int v = 0; v < 8; ++v) {
      int row = rowsS[p * 16 + v + mhalf];
      if (row >= 0)
        out[(long)row * COUT + nt * 16 + ncol] = sinf(OMEGA0 * acc[q][v]);
    }
  }
}

// ---------------- launcher ----------------
extern "C" void kernel_launch(void* const* d_in, const int* in_sizes, int n_in,
                              void* d_out, int out_size, void* d_ws, size_t ws_size,
                              hipStream_t stream) {
  const float* feats  = (const float*)d_in[0];   // [B, 512]
  const float* coords = (const float*)d_in[1];   // [1, B, 2]
  const float* W      = (const float*)d_in[2];   // [4, 512, 512]
  float* out = (float*)d_out;

  char* ws = (char*)d_ws;
  int* cnt  = (int*)(ws + 0);
  int* perm = (int*)(ws + 256);
  bf16_t* Bh = (bf16_t*)(ws + 256 + sizeof(int) * (size_t)NEXP * BATCH);
  bf16_t* Bl = Bh + (size_t)NEXP * COUT * CIN;

  zero_cnt_kernel<<<1, 32, 0, stream>>>(cnt);
  build_perm_kernel<<<BATCH / 256, 256, 0, stream>>>(coords, cnt, perm);
  convert_w_kernel<<<(NEXP * COUT * CIN) / 256, 256, 0, stream>>>(W, Bh, Bl);
  moe_siren_gemm_kernel<<<dim3(BATCH / MTILE, NEXP), TPB, LDS_TOTAL, stream>>>(
      feats, Bh, Bl, perm, cnt, out);
}
